// GATLayer_66726611911053
// MI455X (gfx1250) — matrix operand order repaired and checked by reading, not compiled, not run on hardware
//
#include <hip/hip_runtime.h>

// Problem constants (match reference)
#define NN   50000      // nodes
#define EE   800000     // edges
#define HH   4          // heads
#define DD   64         // per-head dim
#define FDIM 256        // HH*DD = in_dim = out_dim
#define NEG_SLOPE 0.2f

typedef __bf16 bf16;
typedef bf16 v16bf __attribute__((ext_vector_type(16)));
typedef bf16 v8bf  __attribute__((ext_vector_type(8)));
typedef float v8f  __attribute__((ext_vector_type(8)));

// ---------------------------------------------------------------- helpers
__device__ __forceinline__ float lrelu(float x) { return x > 0.f ? x : NEG_SLOPE * x; }

// Float atomic max via integer punning (safe given init = -FLT_MAX):
// positive floats compare correctly as signed ints; negative floats compare
// inversely as unsigned ints.
__device__ __forceinline__ void atomicMaxF(float* addr, float v) {
    if (v >= 0.f) atomicMax((int*)addr, __float_as_int(v));
    else          atomicMin((unsigned int*)addr, __float_as_uint(v));
}

// ---------------------------------------------------------------- init
__global__ void k_init(float* __restrict__ out, float* __restrict__ nmax,
                       float* __restrict__ nsum) {
    int tid = blockIdx.x * blockDim.x + threadIdx.x;
    if (tid < NN * FDIM) out[tid] = 0.f;                  // rst accumulator
    if (tid < NN * HH) { nmax[tid] = -3.402823466e38f; nsum[tid] = 0.f; }
}

// ---------------------------------------------------------------- fp32 -> bf16 (+ W transpose)
__global__ void k_convert(const float* __restrict__ h, const float* __restrict__ W,
                          bf16* __restrict__ hb, bf16* __restrict__ wt) {
    int tid = blockIdx.x * blockDim.x + threadIdx.x;
    if (tid < NN * FDIM) hb[tid] = (bf16)h[tid];
    if (tid < FDIM * FDIM) {
        int n = tid >> 8, k = tid & 255;                  // wt[n][k] = W[k][n]
        wt[tid] = (bf16)W[k * FDIM + n];
    }
}

// ---------------------------------------------------------------- WMMA GEMM + fused el/er
// 128-thread block = 4 waves; block covers a 16x256 row-block of feat.
// Wave w computes the 16x64 tile for head w: 4 accumulators share one A
// fragment (4x A reuse), 32 x v_wmma_f32_16x16x32_bf16 per wave.
// Epilogue: per-(row,head) attn dot products reduced in-wave via shfl_xor
// (owned exclusively by this wave since the tile spans the whole head).
__global__ void __launch_bounds__(128) k_gemm(const bf16* __restrict__ hb,
                                              const bf16* __restrict__ wt,
                                              float* __restrict__ feat,
                                              const float* __restrict__ attn_l,
                                              const float* __restrict__ attn_r,
                                              float* __restrict__ el,
                                              float* __restrict__ er) {
    const int lane = threadIdx.x & 31;
    const int w    = threadIdx.x >> 5;     // wave id == head id (0..3)
    const int m0   = blockIdx.x * 16;
    const int half = lane >> 4;            // 0: lanes 0-15, 1: lanes 16-31
    const int idx  = lane & 15;

    const bf16* arow = hb + (size_t)(m0 + idx) * FDIM;   // A row m = idx
    const bf16* brow[4];
    #pragma unroll
    for (int t = 0; t < 4; ++t)
        brow[t] = wt + (size_t)(w * 64 + t * 16 + idx) * FDIM;   // B col n = ...

    v8f acc[4] = {};
    #pragma unroll
    for (int kb = 0; kb < FDIM; kb += 32) {
        // A 16x32 bf16, this code's k order (B same): lanes 0-15 hold K=0..15, lanes 16-31 hold K=16..31
        v16bf a = *(const v16bf*)(arow + kb + half * 16);
        // B 32x16 bf16, this code's k order (A same): lanes 0-15 hold K=0..15, lanes 16-31 hold K=16..31
        #pragma unroll
        for (int t = 0; t < 4; ++t) {
            v16bf b = *(const v16bf*)(brow[t] + kb + half * 16);
            acc[t] = __builtin_amdgcn_wmma_f32_16x16x32_bf16(
                         false, a, false, b, (short)0, acc[t], false, false);
        }
    }

    // store feat: C/D layout -> M = half*8 + r, N = w*64 + t*16 + idx
    #pragma unroll
    for (int t = 0; t < 4; ++t) {
        float* cp = feat + (size_t)(m0 + half * 8) * FDIM + w * 64 + t * 16 + idx;
        #pragma unroll
        for (int r = 0; r < 8; ++r)
            cp[(size_t)r * FDIM] = acc[t][r];
    }

    // fused el/er: per-lane partial over this lane's 4 columns, then butterfly
    // reduce across the 16 lanes of each half (masks < 16 stay in-half).
    float al[4], ar[4];
    #pragma unroll
    for (int t = 0; t < 4; ++t) {
        al[t] = attn_l[w * 64 + t * 16 + idx];
        ar[t] = attn_r[w * 64 + t * 16 + idx];
    }
    float pel[8], per_[8];
    #pragma unroll
    for (int r = 0; r < 8; ++r) {
        pel[r]  = acc[0][r] * al[0] + acc[1][r] * al[1] + acc[2][r] * al[2] + acc[3][r] * al[3];
        per_[r] = acc[0][r] * ar[0] + acc[1][r] * ar[1] + acc[2][r] * ar[2] + acc[3][r] * ar[3];
    }
    #pragma unroll
    for (int m = 8; m >= 1; m >>= 1) {
        #pragma unroll
        for (int r = 0; r < 8; ++r) {
            pel[r]  += __shfl_xor(pel[r],  m, 32);
            per_[r] += __shfl_xor(per_[r], m, 32);
        }
    }
    if (idx == 0) {
        #pragma unroll
        for (int r = 0; r < 8; ++r) {
            el[(size_t)(m0 + half * 8 + r) * HH + w] = pel[r];
            er[(size_t)(m0 + half * 8 + r) * HH + w] = per_[r];
        }
    }
}

// ---------------------------------------------------------------- edge pass 1: segment max
__global__ void k_edge_max(const int* __restrict__ src, const int* __restrict__ dst,
                           const float* __restrict__ el, const float* __restrict__ er,
                           float* __restrict__ nmax) {
    int e = blockIdx.x * blockDim.x + threadIdx.x;
    if (e >= EE) return;
    int s = src[e], d = dst[e];
    float4 a = *(const float4*)(el + (size_t)s * HH);
    float4 b = *(const float4*)(er + (size_t)d * HH);
    atomicMaxF(nmax + (size_t)d * HH + 0, lrelu(a.x + b.x));
    atomicMaxF(nmax + (size_t)d * HH + 1, lrelu(a.y + b.y));
    atomicMaxF(nmax + (size_t)d * HH + 2, lrelu(a.z + b.z));
    atomicMaxF(nmax + (size_t)d * HH + 3, lrelu(a.w + b.w));
}

// ---------------------------------------------------------------- edge pass 2: segment sum of exp
__global__ void k_edge_sum(const int* __restrict__ src, const int* __restrict__ dst,
                           const float* __restrict__ el, const float* __restrict__ er,
                           const float* __restrict__ nmax, float* __restrict__ nsum) {
    int e = blockIdx.x * blockDim.x + threadIdx.x;
    if (e >= EE) return;
    int s = src[e], d = dst[e];
    float4 a = *(const float4*)(el + (size_t)s * HH);
    float4 b = *(const float4*)(er + (size_t)d * HH);
    float4 m = *(const float4*)(nmax + (size_t)d * HH);
    atomicAdd(nsum + (size_t)d * HH + 0, __expf(lrelu(a.x + b.x) - m.x));
    atomicAdd(nsum + (size_t)d * HH + 1, __expf(lrelu(a.y + b.y) - m.y));
    atomicAdd(nsum + (size_t)d * HH + 2, __expf(lrelu(a.z + b.z) - m.z));
    atomicAdd(nsum + (size_t)d * HH + 3, __expf(lrelu(a.w + b.w) - m.w));
}

// ---------------------------------------------------------------- edge pass 3: weighted aggregate
// One 256-thread block per edge; thread t = h*64 + d. src/dst are block-uniform
// (scalar loads); feat + accumulator are L2-resident (~102 MB < 192 MB L2).
__global__ void __launch_bounds__(256) k_edge_agg(
        const int* __restrict__ src, const int* __restrict__ dst,
        const float* __restrict__ el, const float* __restrict__ er,
        const float* __restrict__ nmax, const float* __restrict__ nsum,
        const float* __restrict__ feat, float* __restrict__ out) {
    int e = blockIdx.x;
    int t = threadIdx.x;               // h*64 + d
    int s = src[e], d = dst[e];
    int hh = t >> 6;
    float x     = lrelu(el[(size_t)s * HH + hh] + er[(size_t)d * HH + hh]);
    float alpha = __expf(x - nmax[(size_t)d * HH + hh]) / nsum[(size_t)d * HH + hh];
    atomicAdd(out + (size_t)d * FDIM + t, feat[(size_t)s * FDIM + t] * alpha);
}

// ---------------------------------------------------------------- epilogue: out = h + elu(rst + bias)
__global__ void k_final(const float* __restrict__ h, const float* __restrict__ bias,
                        float* __restrict__ out) {
    int tid = blockIdx.x * blockDim.x + threadIdx.x;
    if (tid >= NN * FDIM) return;
    float r = out[tid] + bias[tid & (FDIM - 1)];
    r = r > 0.f ? r : (__expf(r) - 1.f);
    out[tid] = r + h[tid];
}

// ---------------------------------------------------------------- launch
extern "C" void kernel_launch(void* const* d_in, const int* in_sizes, int n_in,
                              void* d_out, int out_size, void* d_ws, size_t ws_size,
                              hipStream_t stream) {
    const float* h      = (const float*)d_in[0];
    const float* W      = (const float*)d_in[1];
    const float* attn_l = (const float*)d_in[2];
    const float* attn_r = (const float*)d_in[3];
    const float* bias   = (const float*)d_in[4];
    const int*   src    = (const int*)d_in[5];
    const int*   dst    = (const int*)d_in[6];
    float* out = (float*)d_out;

    // carve workspace (aligned 256B)
    char* p = (char*)d_ws;
    auto carve = [&](size_t bytes) {
        void* r = (void*)p;
        p += (bytes + 255) & ~(size_t)255;
        return r;
    };
    float* feat = (float*)carve((size_t)NN * FDIM * sizeof(float));   // 51.2 MB
    bf16*  hb   = (bf16*) carve((size_t)NN * FDIM * sizeof(bf16));    // 25.6 MB
    bf16*  wt   = (bf16*) carve((size_t)FDIM * FDIM * sizeof(bf16));  // 128 KB
    float* el   = (float*)carve((size_t)NN * HH * sizeof(float));
    float* er   = (float*)carve((size_t)NN * HH * sizeof(float));
    float* nmax = (float*)carve((size_t)NN * HH * sizeof(float));
    float* nsum = (float*)carve((size_t)NN * HH * sizeof(float));

    const int g_nf = (NN * FDIM + 255) / 256;

    k_init   <<<g_nf, 256, 0, stream>>>(out, nmax, nsum);
    k_convert<<<g_nf, 256, 0, stream>>>(h, W, hb, wt);
    k_gemm   <<<NN / 16, 128, 0, stream>>>(hb, wt, feat, attn_l, attn_r, el, er);
    k_edge_max<<<(EE + 255) / 256, 256, 0, stream>>>(src, dst, el, er, nmax);
    k_edge_sum<<<(EE + 255) / 256, 256, 0, stream>>>(src, dst, el, er, nmax, nsum);
    k_edge_agg<<<EE, 256, 0, stream>>>(src, dst, el, er, nmax, nsum, feat, out);
    k_final  <<<g_nf, 256, 0, stream>>>(h, bias, out);
}
